// MultiTypeCategorical_51591147159566
// MI455X (gfx1250) — matrix operand order repaired and checked by reading, not compiled, not run on hardware
//
#include <hip/hip_runtime.h>
#include <hip/hip_bf16.h>
#include <math.h>

typedef __bf16 bf16;
typedef bf16  v16bf __attribute__((ext_vector_type(16)));
typedef float v8f   __attribute__((ext_vector_type(8)));

#define BATCH 4096
#define IN    512
#define OUTD  64
#define NEXP  128
#define NDST  32
#define NTYP  4

// ---------------- Kernel 1: x f32 -> bf16 (coalesced) ----------------
__global__ void cvt_x_kernel(const float* __restrict__ x, bf16* __restrict__ xb, int n) {
    int i = blockIdx.x * 256 + threadIdx.x;
    if (i < n) xb[i] = (bf16)x[i];
}

// ---------------- Kernel 2: W[e][i][o] f32 -> Wt[e][o][i] bf16 (LDS transpose) ----
// grid = NEXP * (IN/64), block = 256
__global__ void cvt_w_kernel(const float* __restrict__ W, bf16* __restrict__ Wt) {
    __shared__ bf16 tile[64][OUTD + 4];   // [i_local][o], padded vs bank conflicts
    int e  = blockIdx.x >> 3;
    int i0 = (blockIdx.x & 7) * 64;
    const float* src = W + (size_t)e * IN * OUTD + (size_t)i0 * OUTD;
    for (int k = threadIdx.x; k < 64 * OUTD; k += 256) {
        int iL = k >> 6, o = k & 63;
        tile[iL][o] = (bf16)src[k];
    }
    __syncthreads();
    // thread t: output row o = t>>2, i-segment seg = (t&3)*16 ; write 16 bf16 = 8 dwords
    int o   = threadIdx.x >> 2;
    int seg = (threadIdx.x & 3) * 16;
    bf16* dst = Wt + ((size_t)e * OUTD + o) * IN + i0 + seg;
    unsigned* d32 = (unsigned*)dst;
    union P { bf16 h[2]; unsigned u; };
#pragma unroll
    for (int j = 0; j < 8; ++j) {
        P p;
        p.h[0] = tile[seg + 2 * j][o];
        p.h[1] = tile[seg + 2 * j + 1][o];
        d32[j] = p.u;
    }
}

// ---------------- Kernel 3: gated expert GEMM + bias + log_softmax ----------------
// grid = BATCH/32, block = 256 (8 waves: 2 M-tiles x 4 N-tiles)
__global__ __launch_bounds__(256, 1)
void moe_gemm_kernel(const float* __restrict__ dst_state,
                     const float* __restrict__ type_state,
                     const float* __restrict__ bias,
                     const bf16* __restrict__ xb,
                     const bf16* __restrict__ Wt,
                     float* __restrict__ out) {
    __shared__ float gls[32][NEXP];       // gating weights for this WG's 32 rows
    __shared__ float slog[32][OUTD + 4];  // logits staging
    __shared__ float rmax[32], rlse[32];

    const int tid = threadIdx.x;
    const int m0  = blockIdx.x * 32;

    // g[b,e] = dst[b, e>>2] * type[b, e&3]
    for (int k = tid; k < 32 * NEXP; k += 256) {
        int r = k >> 7, e = k & 127;
        gls[r][e] = dst_state[(m0 + r) * NDST + (e >> 2)] *
                    type_state[(m0 + r) * NTYP + (e & 3)];
    }
    __syncthreads();

    const int wave  = tid >> 5, lane = tid & 31;
    const int mt    = wave >> 2, nt  = wave & 3;
    const int col   = lane & 15, khalf = lane >> 4;
    const int arow  = m0 + mt * 16 + col;      // A-layout: lanes l and l+16 both hold row M=l
    const int ocol  = nt * 16 + col;           // B/C/D column
    const int rbase = mt * 16 + khalf * 8;     // C/D local row base (VGPR r -> row r + 8*khalf)

    // Hoist all 16 A fragments (K=512) into registers.
    // A 16-bit layout: lane khalf=0 holds K 0-7 & 16-23; khalf=1 holds K 8-15 & 24-31.
    v16bf a[16];
    const bf16* xr = xb + (size_t)arow * IN;
#pragma unroll
    for (int ib = 0; ib < 16; ++ib) {
        const bf16* p = xr + ib * 32 + khalf * 8;
        union { v16bf v; uint4 q[2]; } u;
        u.q[0] = *(const uint4*)p;           // K local khalf*8 .. +7
        u.q[1] = *(const uint4*)(p + 16);    // K local khalf*8+16 .. +23
        a[ib] = u.v;
    }

    v8f logit = {};
    const bf16* wbase = Wt + (size_t)ocol * IN + khalf * 8;   // B: lanes0-15 K0-7 & 16-23, lanes16-31 K8-15 & 24-31

    for (int e = 0; e < NEXP; ++e) {
        v8f acc = {};
        const bf16* we = wbase + (size_t)e * OUTD * IN;
#pragma unroll
        for (int ib = 0; ib < 16; ++ib) {
            const bf16* q = we + ib * 32;
            union { v16bf v; uint4 qq[2]; } u;
            u.qq[0] = *(const uint4*)q;          // K local khalf*8 .. +7
            u.qq[1] = *(const uint4*)(q + 16);
            acc = __builtin_amdgcn_wmma_f32_16x16x32_bf16(
                false, a[ib], false, u.v, (short)0, acc, false, false);
        }
        const float bv = bias[e * OUTD + ocol];
#pragma unroll
        for (int r = 0; r < 8; ++r) {
            logit[r] += gls[rbase + r][e] * (acc[r] + bv);   // broadcast LDS reads
        }
    }

#pragma unroll
    for (int r = 0; r < 8; ++r) slog[rbase + r][ocol] = logit[r];
    __syncthreads();

    if (tid < 32) {
        float m = -INFINITY;
        for (int o = 0; o < OUTD; ++o) m = fmaxf(m, slog[tid][o]);
        float s = 0.f;
        for (int o = 0; o < OUTD; ++o) s += __expf(slog[tid][o] - m);
        rmax[tid] = m;
        rlse[tid] = __logf(s);
    }
    __syncthreads();

    for (int k = tid; k < 32 * OUTD; k += 256) {
        int r = k >> 6, o = k & 63;
        out[(size_t)(m0 + r) * OUTD + o] = slog[r][o] - rmax[r] - rlse[r];
    }
}

extern "C" void kernel_launch(void* const* d_in, const int* in_sizes, int n_in,
                              void* d_out, int out_size, void* d_ws, size_t ws_size,
                              hipStream_t stream) {
    const float* x   = (const float*)d_in[0];   // [4096,512]
    const float* dst = (const float*)d_in[1];   // [4096,32]
    const float* typ = (const float*)d_in[2];   // [4096,4]
    const float* W   = (const float*)d_in[3];   // [128,512,64]
    const float* b   = (const float*)d_in[4];   // [128,64]
    float* out = (float*)d_out;                 // [4096,64]

    bf16* xb = (bf16*)d_ws;                                            // 4 MB
    bf16* Wt = (bf16*)((char*)d_ws + (size_t)BATCH * IN * sizeof(bf16)); // 8.4 MB

    int nx = BATCH * IN;
    cvt_x_kernel<<<(nx + 255) / 256, 256, 0, stream>>>(x, xb, nx);
    cvt_w_kernel<<<NEXP * (IN / 64), 256, 0, stream>>>(W, Wt);
    moe_gemm_kernel<<<BATCH / 32, 256, 0, stream>>>(dst, typ, b, xb, Wt, out);
}
